// BondDecoder_44873818309252
// MI455X (gfx1250) — matrix operand order repaired and checked by reading, not compiled, not run on hardware
//
#include <hip/hip_runtime.h>

// ---------------------------------------------------------------------------
// BondDecoder loss for MI455X (gfx1250, wave32, WMMA).
//
// Shapes: L=512, B=64, D=256, H=4 heads, hd=64, MAX_BONDS=6.
//
// Algebraic fusion: pointwise Conv1d + MHA q/k projection compose into one
// linear per tensor:  q = x (W2 W1)^T + (W2 b1 + b2), with the 1/sqrt(hd)
// scale folded into the q-side weights. Halves projection GEMM work.
//
// Stage 1: project   — 4 GEMMs [32768x256]x[256x256] in f16 WMMA -> Q/K (f16)
// Stage 2: attn_loss — per (b, 16-row q-tile): WMMA scores per head/branch
//          into LDS, masked softmax (wave32 shuffle reductions), head mean,
//          fused (inc-dec)*4 + bond-one-hot squared-error reduction,
//          global f32 atomic accumulation into loss[b].
// Attention maps (2 x 64 MB if materialized) never leave LDS (320 KB/WGP).
// global_prefetch_b8 pulls the next head's K slice while WMMAs run.
// ---------------------------------------------------------------------------

typedef __attribute__((ext_vector_type(16))) _Float16 v16h;
typedef __attribute__((ext_vector_type(8)))  float    v8f;
typedef __attribute__((ext_vector_type(4)))  float    f4;

union ABfrag { v16h v; f4 q[2]; };

__device__ __forceinline__ v8f wmma_f16(v16h a, v16h b, v8f c) {
    return __builtin_amdgcn_wmma_f32_16x16x32_f16(false, a, false, b,
                                                  (short)0, c, false, false);
}

static constexpr int Lq = 512;
static constexpr int Bq = 64;
static constexpr int Dq = 256;
static constexpr int NB = 6;
static constexpr size_t QKSZ = (size_t)Bq * Lq * Dq;   // halves per matrix

// ---------------------------------------------------------------------------
__global__ void zero_out(float* out) {
    if (threadIdx.x < Bq) out[threadIdx.x] = 0.0f;
}

// ---------------------------------------------------------------------------
// Wc[sel][f][d] = scale * sum_e W2[f,e] * W1[e,d]   (f16 output)
// bc[sel][f]    = scale * (sum_e W2[f,e]*b1[e] + b2[f])
// sel: 0=inc_q (scale 1/8), 1=inc_k, 2=dec_q (1/8), 3=dec_k
__global__ void combine_weights(
    const float* iqw, const float* iqb, const float* ikw, const float* ikb,
    const float* dqw, const float* dqb, const float* dkw, const float* dkb,
    const float* iwq, const float* ibq, const float* iwk, const float* ibk,
    const float* dwq, const float* dbq, const float* dwk, const float* dbk,
    _Float16* Wc, float* bc)
{
    const int sel = blockIdx.y;
    const float *W1, *B1, *W2, *B2; float scale;
    switch (sel) {
        case 0:  W1 = iqw; B1 = iqb; W2 = iwq; B2 = ibq; scale = 0.125f; break;
        case 1:  W1 = ikw; B1 = ikb; W2 = iwk; B2 = ibk; scale = 1.0f;   break;
        case 2:  W1 = dqw; B1 = dqb; W2 = dwq; B2 = dbq; scale = 0.125f; break;
        default: W1 = dkw; B1 = dkb; W2 = dwk; B2 = dbk; scale = 1.0f;   break;
    }
    const int f = blockIdx.x;
    const int d = threadIdx.x;
    float acc = 0.0f;
    for (int e = 0; e < Dq; ++e)
        acc += W2[f * Dq + e] * W1[e * Dq + d];
    Wc[((size_t)sel * Dq + f) * Dq + d] = (_Float16)(acc * scale);
    if (d == 0) {
        float bacc = 0.0f;
        for (int e = 0; e < Dq; ++e) bacc += W2[f * Dq + e] * B1[e];
        bc[sel * Dq + f] = (bacc + B2[f]) * scale;
    }
}

// ---------------------------------------------------------------------------
// QK[sel][b][l][f] = sum_d X[l][b][d] * Wc[sel][f][d] + bc[sel][f]   (f16)
// X rows indexed m = l*B + b (contiguous D floats). 8 waves/block, each wave
// owns a 16x32 output strip; K loop of 8 x wmma_f32_16x16x32_f16.
__global__ __launch_bounds__(256) void project(
    const float* __restrict__ X, const _Float16* __restrict__ Wc,
    const float* __restrict__ bc, _Float16* __restrict__ QK)
{
    const int sel  = blockIdx.y;
    const int m0   = blockIdx.x * 16;
    const int tid  = threadIdx.x;
    const int lane = tid & 31;
    const int wave = tid >> 5;
    const int n0   = wave * 32;

    const _Float16* Wsel = Wc + (size_t)sel * Dq * Dq;
    const float*    bsel = bc + sel * Dq;
    _Float16*       out  = QK + (size_t)sel * QKSZ;

    const int r     = lane & 15;
    const int aksel = (lane & 16) ? 8 : 0;    // A frag: K sub-offset
    const int bksel = (lane & 16) ? 8 : 0;    // B frag: K sub-offset

    const float*    arow  = X + (size_t)(m0 + r) * Dq;
    const _Float16* brow0 = Wsel + (size_t)(n0 + r) * Dq;
    const _Float16* brow1 = Wsel + (size_t)(n0 + 16 + r) * Dq;

    v8f c0 = {}, c1 = {};
    #pragma unroll
    for (int k0 = 0; k0 < Dq; k0 += 32) {
        if (k0 + 64 < Dq) {
            // pull the next K-chunk toward the WGP while WMMAs run
            __builtin_prefetch(arow + k0 + 64, 0, 1);
            __builtin_prefetch(brow0 + k0 + 64, 0, 1);
            __builtin_prefetch(brow1 + k0 + 64, 0, 1);
        }
        ABfrag a;
        f4 x0 = *(const f4*)(arow + k0 + aksel);
        f4 x1 = *(const f4*)(arow + k0 + aksel + 4);
        f4 x2 = *(const f4*)(arow + k0 + 16 + aksel);
        f4 x3 = *(const f4*)(arow + k0 + 16 + aksel + 4);
        #pragma unroll
        for (int t = 0; t < 4; ++t) {
            a.v[t]      = (_Float16)x0[t];
            a.v[4 + t]  = (_Float16)x1[t];
            a.v[8 + t]  = (_Float16)x2[t];
            a.v[12 + t] = (_Float16)x3[t];
        }
        ABfrag b0, b1;
        b0.q[0] = *(const f4*)(brow0 + k0 + bksel);
        b0.q[1] = *(const f4*)(brow0 + k0 + bksel + 16);
        b1.q[0] = *(const f4*)(brow1 + k0 + bksel);
        b1.q[1] = *(const f4*)(brow1 + k0 + bksel + 16);
        c0 = wmma_f16(a.v, b0.v, c0);
        c1 = wmma_f16(a.v, b1.v, c1);
    }

    const int row = (lane & 16) ? 8 : 0;
    const int col = lane & 15;
    #pragma unroll
    for (int g = 0; g < 8; ++g) {
        int m  = m0 + row + g;
        int l  = m >> 6;          // B = 64
        int bb = m & 63;
        size_t o = ((size_t)(bb * Lq + l)) * Dq;
        out[o + n0 + col]      = (_Float16)(c0[g] + bsel[n0 + col]);
        out[o + n0 + 16 + col] = (_Float16)(c1[g] + bsel[n0 + 16 + col]);
    }
}

// ---------------------------------------------------------------------------
// One block per (b, 16-row q-tile). Scores per head/branch via WMMA into LDS,
// masked softmax + head-mean in LDS, fused bond/mask loss, atomic into out[b].
__global__ __launch_bounds__(256) void attn_loss(
    const _Float16* __restrict__ QK,
    const int* __restrict__ src_bond, const unsigned char* __restrict__ src_mask,
    const int* __restrict__ tgt_bond, const unsigned char* __restrict__ tgt_mask,
    float* __restrict__ out)
{
    __shared__ float sc[16][Lq];        // 32 KB: scores of current head
    __shared__ float avI[16][Lq];       // 32 KB: head-averaged inc probs
    __shared__ float avD[16][Lq];       // 32 KB: head-averaged dec probs
    __shared__ unsigned char smask[Lq];
    __shared__ unsigned char tmask[Lq];
    __shared__ int sbond[16 * NB];
    __shared__ int tbond[16 * NB];
    __shared__ float wsum[8];

    const int b    = blockIdx.y;
    const int i0   = blockIdx.x * 16;
    const int tid  = threadIdx.x;
    const int lane = tid & 31;
    const int wave = tid >> 5;

    for (int j = tid; j < Lq; j += 256) {
        smask[j] = src_mask[b * Lq + j];
        tmask[j] = tgt_mask[b * Lq + j];
    }
    if (tid < 16 * NB) {
        int i = i0 + tid / NB, nb = tid % NB;
        sbond[tid] = src_bond[((size_t)b * Lq + i) * NB + nb];
        tbond[tid] = tgt_bond[((size_t)b * Lq + i) * NB + nb];
    }
    for (int x = tid; x < 16 * Lq; x += 256) {
        (&avI[0][0])[x] = 0.0f;
        (&avD[0][0])[x] = 0.0f;
    }
    __syncthreads();

    const int r     = lane & 15;
    const int aksel = (lane & 16) ? 8 : 0;
    const int bksel = (lane & 16) ? 8 : 0;
    const int row   = (lane & 16) ? 8 : 0;
    const int col   = lane & 15;

    for (int br = 0; br < 2; ++br) {
        const _Float16* qb = QK + (size_t)(2 * br) * QKSZ
                           + ((size_t)(b * Lq + i0 + r)) * Dq;
        const _Float16* kbase = QK + (size_t)(2 * br + 1) * QKSZ
                              + ((size_t)b * Lq) * Dq;
        float (*avg)[Lq] = br ? avD : avI;

        for (int h = 0; h < 4; ++h) {
            const int hb = h * 64;
            // Q (A-matrix) fragments: invariant across j-tiles, load once.
            ABfrag a0, a1;
            a0.q[0] = *(const f4*)(qb + hb + aksel);
            a0.q[1] = *(const f4*)(qb + hb + 16 + aksel);
            a1.q[0] = *(const f4*)(qb + hb + 32 + aksel);
            a1.q[1] = *(const f4*)(qb + hb + 48 + aksel);

            // --- scores[i0..i0+15][0..511] for this head -------------------
            #pragma unroll
            for (int t = 0; t < 4; ++t) {
                const int j0 = (wave * 4 + t) * 16;
                const _Float16* kb = kbase + (size_t)(j0 + r) * Dq;
                if (h < 3) __builtin_prefetch(kb + hb + 64, 0, 1);
                ABfrag b0, b1;
                b0.q[0] = *(const f4*)(kb + hb + bksel);
                b0.q[1] = *(const f4*)(kb + hb + bksel + 16);
                b1.q[0] = *(const f4*)(kb + hb + 32 + bksel);
                b1.q[1] = *(const f4*)(kb + hb + 32 + bksel + 16);
                v8f c = {};
                c = wmma_f16(a0.v, b0.v, c);
                c = wmma_f16(a1.v, b1.v, c);
                #pragma unroll
                for (int g = 0; g < 8; ++g) sc[row + g][j0 + col] = c[g];
            }
            __syncthreads();

            // --- masked softmax over j, accumulate head mean ---------------
            #pragma unroll
            for (int rr = 0; rr < 2; ++rr) {
                const int rw = wave * 2 + rr;
                float m = -3.0e38f;
                for (int j = lane; j < Lq; j += 32) {
                    float s = smask[j] ? -1.0e30f : sc[rw][j];
                    m = fmaxf(m, s);
                }
                #pragma unroll
                for (int o = 16; o; o >>= 1) m = fmaxf(m, __shfl_xor(m, o, 32));
                float sum = 0.0f;
                for (int j = lane; j < Lq; j += 32) {
                    float s = smask[j] ? -1.0e30f : sc[rw][j];
                    float e = __expf(s - m);
                    sc[rw][j] = e;
                    sum += e;
                }
                #pragma unroll
                for (int o = 16; o; o >>= 1) sum += __shfl_xor(sum, o, 32);
                const float inv = 0.25f / sum;    // mean over 4 heads
                for (int j = lane; j < Lq; j += 32)
                    avg[rw][j] += sc[rw][j] * inv;
            }
            __syncthreads();
        }
    }

    // --- fused loss: ((inc-dec)*4 + src_onehot - tgt_onehot)^2 * gates -----
    float part = 0.0f;
    for (int x = tid; x < 16 * Lq; x += 256) {
        const int il = x >> 9;
        const int j  = x & (Lq - 1);
        const int i  = i0 + il;
        if (!smask[i] && !smask[j]) {                 // pad_mask gate
            const int ti = tmask[i], tj = tmask[j];
            if (!(ti && tj)) {                        // or_mask gate
                float e = (avI[il][j] - avD[il][j]) * 4.0f;
                int sw = 0;
                #pragma unroll
                for (int nb = 0; nb < NB; ++nb) sw += (sbond[il * NB + nb] == j);
                e += (float)sw;
                if (!ti && !tj) {                     // and_mask gate
                    int tw = 0;
                    #pragma unroll
                    for (int nb = 0; nb < NB; ++nb) tw += (tbond[il * NB + nb] == j);
                    e -= (float)tw;
                }
                part += e * e;
            }
        }
    }
    #pragma unroll
    for (int o = 16; o; o >>= 1) part += __shfl_xor(part, o, 32);
    if (lane == 0) wsum[wave] = part;
    __syncthreads();
    if (tid == 0) {
        float t = 0.0f;
        #pragma unroll
        for (int w = 0; w < 8; ++w) t += wsum[w];
        atomicAdd(out + b, t);
    }
}

// ---------------------------------------------------------------------------
extern "C" void kernel_launch(void* const* d_in, const int* in_sizes, int n_in,
                              void* d_out, int out_size, void* d_ws, size_t ws_size,
                              hipStream_t stream)
{
    (void)in_sizes; (void)n_in; (void)out_size; (void)ws_size;

    const float* X   = (const float*)d_in[0];
    const float* iqw = (const float*)d_in[1];
    const float* iqb = (const float*)d_in[2];
    const float* ikw = (const float*)d_in[3];
    const float* ikb = (const float*)d_in[4];
    const float* dqw = (const float*)d_in[5];
    const float* dqb = (const float*)d_in[6];
    const float* dkw = (const float*)d_in[7];
    const float* dkb = (const float*)d_in[8];
    const float* iwq = (const float*)d_in[9];
    const float* ibq = (const float*)d_in[10];
    const float* iwk = (const float*)d_in[11];
    const float* ibk = (const float*)d_in[12];
    const float* dwq = (const float*)d_in[13];
    const float* dbq = (const float*)d_in[14];
    const float* dwk = (const float*)d_in[15];
    const float* dbk = (const float*)d_in[16];
    const int*           sbond = (const int*)d_in[17];
    const unsigned char* smask = (const unsigned char*)d_in[18];
    const int*           tbond = (const int*)d_in[19];
    const unsigned char* tmask = (const unsigned char*)d_in[20];
    float* out = (float*)d_out;

    // workspace layout
    char* ws = (char*)d_ws;
    _Float16* Wc = (_Float16*)ws;                               // 4*256*256*2 B
    float*    bc = (float*)(ws + (size_t)4 * Dq * Dq * 2);      // 4*256*4 B
    _Float16* QK = (_Float16*)(ws + (size_t)4 * Dq * Dq * 2 + 4 * Dq * 4);

    zero_out<<<1, 64, 0, stream>>>(out);
    combine_weights<<<dim3(Dq, 4), Dq, 0, stream>>>(
        iqw, iqb, ikw, ikb, dqw, dqb, dkw, dkb,
        iwq, ibq, iwk, ibk, dwq, dbq, dwk, dbk, Wc, bc);
    project<<<dim3((Bq * Lq) / 16, 4), 256, 0, stream>>>(X, Wc, bc, QK);
    attn_loss<<<dim3(Lq / 16, Bq), 256, 0, stream>>>(QK, sbond, smask,
                                                     tbond, tmask, out);
}
